// FoldingDecoderSphereTwoShellAttn_2430951489910
// MI455X (gfx1250) — compile-verified, matrix operand order checked by reading, not run on hardware
//
#include <hip/hip_runtime.h>
#include <math.h>

// ---------------------------------------------------------------- types
typedef __attribute__((ext_vector_type(16))) __bf16 v16bf;
typedef __attribute__((ext_vector_type(8)))  float  v8f;
typedef __attribute__((ext_vector_type(4)))  unsigned int u32x4;
typedef __attribute__((ext_vector_type(8)))  int  i32x8;
typedef __attribute__((ext_vector_type(4)))  int  i32x4;

#define M_TOT   32768      // B*N points
#define NPTS    4096
#define BATCH   8
#define HID     512
#define KPAD    672        // padded concat width (515/643 -> 672)
#define AD      128
#define EPSV    1e-5f
#define RATT2   0.04f      // R_ATT^2
#define INV_SQRT_D 0.08838834764831845f  // 1/sqrt(128)

#define LROW    40                     // LDS row stride in bf16 (64B data + 16B TDM pad)
#define BUFSZ   ((128 + 64) * LROW)    // one double-buffer slot (A tile + W tile)

#if __has_builtin(__builtin_amdgcn_tensor_load_to_lds)
#define HAVE_TDM 1
#else
#define HAVE_TDM 0
#endif

static __device__ inline v8f wmma_bf16(v16bf a, v16bf b, v8f c) {
  return __builtin_amdgcn_wmma_f32_16x16x32_bf16(false, a, false, b, (short)0, c,
                                                 false, false);
}

static __device__ inline unsigned lds_off(const void* p) {
  // flat LDS aperture: low 32 bits hold the LDS byte offset
  return (unsigned)(unsigned long long)(uintptr_t)p;
}

#if HAVE_TDM
// 2-D TDM load: tile (tileD0 x tileD1) of bf16 from row-major tensor, with
// 16B LDS padding after each 64B row (-> LROW stride, bank-conflict free).
static __device__ inline void tdm_load_2d(unsigned ldsAddr, const void* gaddr,
                                          unsigned tensorD0, unsigned tensorD1,
                                          unsigned tileD0, unsigned tileD1,
                                          unsigned long long stride0) {
  unsigned long long ga = (unsigned long long)(uintptr_t)gaddr;
  u32x4 g0;
  g0.x = 1u;                                        // count=1, user mode
  g0.y = ldsAddr;                                   // lds_addr [63:32]
  g0.z = (unsigned)ga;                              // global_addr [95:64]
  g0.w = (unsigned)((ga >> 32) & 0x01FFFFFFu)       // global_addr [120:96]
       | (2u << 30);                                // type=2 (image)
  unsigned w0 = (1u << 16)                          // data_size = 2 bytes
              | (1u << 20)                          // pad_enable
              | (3u << 22)                          // pad_interval = 16 dwords (64B)
              | (3u << 25);                         // pad_amount  = 4 dwords (16B)
  i32x8 g1;
  g1[0] = (int)w0;
  g1[1] = (int)((tensorD0 & 0xFFFFu) << 16);                          // dim0 lo16
  g1[2] = (int)(((tensorD0 >> 16) & 0xFFFFu) | ((tensorD1 & 0xFFFFu) << 16));
  g1[3] = (int)(((tensorD1 >> 16) & 0xFFFFu) | ((tileD0 & 0xFFFFu) << 16));
  g1[4] = (int)(tileD1 & 0xFFFFu);                                    // tile_dim2=0
  g1[5] = (int)(unsigned)(stride0 & 0xFFFFFFFFull);
  g1[6] = (int)(unsigned)((stride0 >> 32) & 0xFFFFull);               // dim1_stride=0
  g1[7] = 0;
  i32x4 zz = {0, 0, 0, 0};
#if defined(__clang_major__) && __clang_major__ >= 23
  i32x8 z8 = {0, 0, 0, 0, 0, 0, 0, 0};
  __builtin_amdgcn_tensor_load_to_lds(g0, g1, zz, zz, z8, 0);
#else
  __builtin_amdgcn_tensor_load_to_lds(g0, g1, zz, zz, 0);
#endif
}
#endif

// ------------------------------------------------ weight pack (f32 -> bf16, K padded)
__global__ __launch_bounds__(256)
void pack_w(const float* __restrict__ src, __bf16* __restrict__ dst,
            int N, int Kin, int Kpad) {
  int idx = blockIdx.x * 256 + threadIdx.x;
  if (idx >= N * Kpad) return;
  int n = idx / Kpad, k = idx - n * Kpad;
  dst[idx] = (__bf16)((k < Kin) ? src[n * Kin + k] : 0.0f);
}

// ------------------------------------------------ X0 = [z | template | zeros]
__global__ __launch_bounds__(256)
void build_x0(const float* __restrict__ z, const float* __restrict__ tmpl,
              __bf16* __restrict__ X) {
  int m = blockIdx.x;
  int b = m >> 12;
  int n = m & 4095;
  __bf16* xr = X + (size_t)m * KPAD;
  for (int k = threadIdx.x; k < KPAD; k += 256) {
    float v;
    if (k < 512)      v = z[b * 512 + k];
    else if (k < 515) v = tmpl[n * 3 + (k - 512)];
    else              v = 0.0f;
    xr[k] = (__bf16)v;
  }
}

// ------------------------------------------------ TDM-fed bf16 GEMM: C = A * W^T + bias
// A: (M,K) bf16 stride lda ; W: (N,K) bf16 stride ldw ; C: (M,N) f32 stride ldc
// grid = (M/128, N/64), block = 256 (8 waves). Wave w computes rows w*16..+15 x 64 cols.
// Double-buffered LDS tiles filled by the Tensor Data Mover (wave 0 issues,
// s_wait_tensorcnt + barrier to release), compute overlaps the next prefetch.
__global__ __launch_bounds__(256)
void gemm_tdm(const __bf16* __restrict__ A, int lda,
              const __bf16* __restrict__ W, int ldw,
              const float* __restrict__ bias,
              float* __restrict__ C, int ldc, int K) {
  __shared__ __align__(64) __bf16 sm[2 * BUFSZ];   // 30 KB
  int wave = threadIdx.x >> 5;
  int lane = threadIdx.x & 31;
  int half = lane >> 4, l15 = lane & 15;
  int rowBase = blockIdx.x * 128;
  int colBase = blockIdx.y * 64;

  __bf16* sbase = &sm[0];          // buf b: A tile at b*BUFSZ, W tile at +128*LROW
  int KT = K / 32;

#if HAVE_TDM
  if (wave == 0) {   // prefetch k-tile 0 into buffer 0
    tdm_load_2d(lds_off(sbase), A + (size_t)rowBase * lda, (unsigned)K, M_TOT,
                32, 128, (unsigned long long)lda);
    tdm_load_2d(lds_off(sbase + 128 * LROW), W + (size_t)colBase * ldw,
                (unsigned)K, HID, 32, 64, (unsigned long long)ldw);
  }
#endif

  v8f acc[4] = {};
  for (int kt = 0; kt < KT; ++kt) {
#if HAVE_TDM
    int buf = kt & 1;
    __bf16* sA = sbase + buf * BUFSZ;
    __bf16* sW = sA + 128 * LROW;
    if (wave == 0) {
      if (kt + 1 < KT) {      // prefetch next tile into the other buffer
        size_t k2 = (size_t)(kt + 1) * 32;
        __bf16* nA = sbase + (buf ^ 1) * BUFSZ;
        tdm_load_2d(lds_off(nA), A + (size_t)rowBase * lda + k2,
                    (unsigned)K, M_TOT, 32, 128, (unsigned long long)lda);
        tdm_load_2d(lds_off(nA + 128 * LROW), W + (size_t)colBase * ldw + k2,
                    (unsigned)K, HID, 32, 64, (unsigned long long)ldw);
        __builtin_amdgcn_s_wait_tensorcnt(2);   // current tile's pair complete
      } else {
        __builtin_amdgcn_s_wait_tensorcnt(0);
      }
    }
#else
    __bf16* sA = sbase;
    __bf16* sW = sbase + 128 * LROW;
    {   // cooperative fallback copy (no TDM builtin on this toolchain)
      int k2 = kt * 32;
      {
        int r = threadIdx.x >> 1, h = threadIdx.x & 1;
        *(v16bf*)(sA + r * LROW + h * 16) =
            *(const v16bf*)(A + (size_t)(rowBase + r) * lda + k2 + h * 16);
      }
      if (threadIdx.x < 128) {
        int r = threadIdx.x >> 1, h = threadIdx.x & 1;
        *(v16bf*)(sW + r * LROW + h * 16) =
            *(const v16bf*)(W + (size_t)(colBase + r) * ldw + k2 + h * 16);
      }
    }
#endif
    __syncthreads();
    v16bf af = *(const v16bf*)(sA + (wave * 16 + l15) * LROW + half * 16);
#pragma unroll
    for (int j = 0; j < 4; ++j) {
      v16bf bf = *(const v16bf*)(sW + (j * 16 + l15) * LROW + half * 16);
      acc[j] = wmma_bf16(af, bf, acc[j]);
    }
    __syncthreads();
  }

#pragma unroll
  for (int j = 0; j < 4; ++j) {
    int col = colBase + j * 16 + l15;
    float bb = bias ? bias[col] : 0.0f;
#pragma unroll
    for (int v = 0; v < 8; ++v) {
      int row = rowBase + wave * 16 + v + 8 * half;
      C[(size_t)row * ldc + col] = acc[j][v] + bb;
    }
  }
}

// ------------------------------------------------ BN stats, stage 1 (deterministic tree)
__global__ __launch_bounds__(256)
void bn_stats1(const float* __restrict__ H, float* __restrict__ pSum,
               float* __restrict__ pSq) {
  int t  = threadIdx.x;
  int rb = blockIdx.x * 128;
  int c0 = t, c1 = t + 256;
  float s0 = 0, s1 = 0, q0 = 0, q1 = 0;
  for (int r = 0; r < 128; ++r) {
    const float* row = H + (size_t)(rb + r) * HID;
    float x0 = row[c0], x1 = row[c1];
    s0 += x0; q0 += x0 * x0;
    s1 += x1; q1 += x1 * x1;
  }
  pSum[blockIdx.x * HID + c0] = s0;  pSum[blockIdx.x * HID + c1] = s1;
  pSq [blockIdx.x * HID + c0] = q0;  pSq [blockIdx.x * HID + c1] = q1;
}

__global__ __launch_bounds__(256)
void bn_stats2(const float* __restrict__ pSum, const float* __restrict__ pSq,
               const float* __restrict__ g, const float* __restrict__ be,
               float* __restrict__ ab) {
  int c = blockIdx.x * 256 + threadIdx.x;
  float s = 0, q = 0;
  for (int j = 0; j < 256; ++j) { s += pSum[j * HID + c]; q += pSq[j * HID + c]; }
  float mean = s * (1.0f / (float)M_TOT);
  float var  = q * (1.0f / (float)M_TOT) - mean * mean;
  float rstd = rsqrtf(var + EPSV);
  float a = g[c] * rstd;
  ab[c]       = a;
  ab[HID + c] = be[c] - mean * a;
}

__global__ __launch_bounds__(256)
void bn_relu(const float* __restrict__ H, const float* __restrict__ ab,
             __bf16* __restrict__ Hb) {
  int i = (blockIdx.x * 256 + threadIdx.x) * 4;
#pragma unroll
  for (int t = 0; t < 4; ++t) {
    int idx = i + t;
    int c = idx & (HID - 1);
    float y = H[idx] * ab[c] + ab[HID + c];
    Hb[idx] = (__bf16)fmaxf(y, 0.0f);
  }
}

// ------------------------------------------------ 512 -> 3 head (+ |c|^2, + bf16 into X)
__global__ __launch_bounds__(256)
void head3(const __bf16* __restrict__ Hb, const float* __restrict__ w3,
           const float* __restrict__ b3, float* __restrict__ coords,
           float* __restrict__ sq, __bf16* __restrict__ Xn) {
  __shared__ float w[3 * HID];
  for (int i = threadIdx.x; i < 3 * HID; i += 256) w[i] = w3[i];
  __syncthreads();
  int m = blockIdx.x * 256 + threadIdx.x;
  const __bf16* h = Hb + (size_t)m * HID;
  float a0 = b3[0], a1 = b3[1], a2 = b3[2];
  for (int k = 0; k < HID; ++k) {
    float x = (float)h[k];
    a0 += x * w[k];
    a1 += x * w[HID + k];
    a2 += x * w[2 * HID + k];
  }
  coords[m * 3 + 0] = a0;
  coords[m * 3 + 1] = a1;
  coords[m * 3 + 2] = a2;
  if (sq) sq[m] = a0 * a0 + a1 * a1 + a2 * a2;
  if (Xn) {
    __bf16* xr = Xn + (size_t)m * KPAD + 512;
    xr[0] = (__bf16)a0; xr[1] = (__bf16)a1; xr[2] = (__bf16)a2;
  }
}

// ------------------------------------------------ QKV projection (Q prescaled)
__global__ __launch_bounds__(128)
void qkv(const float* __restrict__ coords, const float* __restrict__ wq,
         const float* __restrict__ wk, const float* __restrict__ wv,
         __bf16* __restrict__ Qb, __bf16* __restrict__ Kb, __bf16* __restrict__ Vb) {
  int m = blockIdx.x;
  int d = threadIdx.x;
  float c0 = coords[m * 3 + 0], c1 = coords[m * 3 + 1], c2 = coords[m * 3 + 2];
  float q = c0 * wq[d * 3] + c1 * wq[d * 3 + 1] + c2 * wq[d * 3 + 2];
  float k = c0 * wk[d * 3] + c1 * wk[d * 3 + 1] + c2 * wk[d * 3 + 2];
  float v = c0 * wv[d * 3] + c1 * wv[d * 3 + 1] + c2 * wv[d * 3 + 2];
  size_t o = (size_t)m * AD + d;
  Qb[o] = (__bf16)(q * INV_SQRT_D);
  Kb[o] = (__bf16)k;
  Vb[o] = (__bf16)v;
}

// ------------------------------------------------ flash-style masked local attention
__global__ __launch_bounds__(32)
void attn(const __bf16* __restrict__ Qb, const __bf16* __restrict__ Kb,
          const __bf16* __restrict__ Vb, const float* __restrict__ coords,
          const float* __restrict__ sq, __bf16* __restrict__ Xout) {
  __shared__ __align__(64) __bf16 Vt[AD][32];
  __shared__ __align__(64) __bf16 Ps[16][32];
  __shared__ float  qc[16][3], qs[16], kc[32][3], ks[32];

  int b    = blockIdx.y;
  int qt   = blockIdx.x;
  int lane = threadIdx.x;
  int half = lane >> 4;
  int l15  = lane & 15;
  size_t base = (size_t)b * NPTS;

  const __bf16* qrow = Qb + (base + qt * 16 + l15) * AD + half * 16;
  v16bf aq[4];
#pragma unroll
  for (int c = 0; c < 4; ++c) aq[c] = *(const v16bf*)(qrow + c * 32);

  if (lane < 16) {
    int qi = qt * 16 + lane;
    qc[lane][0] = coords[(base + qi) * 3 + 0];
    qc[lane][1] = coords[(base + qi) * 3 + 1];
    qc[lane][2] = coords[(base + qi) * 3 + 2];
    qs[lane]    = sq[base + qi];
  }
  __syncthreads();

  float mrow[8], lrow[8];
  v8f acc[8] = {};
#pragma unroll
  for (int v = 0; v < 8; ++v) { mrow[v] = -1e30f; lrow[v] = 0.0f; }

  for (int kt = 0; kt < NPTS / 32; ++kt) {
    int k0 = kt * 32;
    const __bf16* vrow = Vb + (base + k0 + lane) * AD;
#pragma unroll 8
    for (int n = 0; n < AD; ++n) Vt[n][lane] = vrow[n];
    kc[lane][0] = coords[(base + k0 + lane) * 3 + 0];
    kc[lane][1] = coords[(base + k0 + lane) * 3 + 1];
    kc[lane][2] = coords[(base + k0 + lane) * 3 + 2];
    ks[lane]    = sq[base + k0 + lane];
    __syncthreads();

    v8f s0 = {}, s1 = {};
#pragma unroll
    for (int c = 0; c < 4; ++c) {
      const __bf16* kr0 = Kb + (base + k0 + l15) * AD + half * 16 + c * 32;
      const __bf16* kr1 = Kb + (base + k0 + 16 + l15) * AD + half * 16 + c * 32;
      v16bf bk0 = *(const v16bf*)kr0;
      v16bf bk1 = *(const v16bf*)kr1;
      s0 = wmma_bf16(aq[c], bk0, s0);
      s1 = wmma_bf16(aq[c], bk1, s1);
    }

    float k0x = kc[l15][0],      k0y = kc[l15][1],      k0z = kc[l15][2],      ks0 = ks[l15];
    float k1x = kc[16 + l15][0], k1y = kc[16 + l15][1], k1z = kc[16 + l15][2], ks1 = ks[16 + l15];
#pragma unroll
    for (int v = 0; v < 8; ++v) {
      int row = v + 8 * half;
      float qx = qc[row][0], qy = qc[row][1], qz = qc[row][2], qsv = qs[row];
      float d2a = qsv + ks0 - 2.0f * (qx * k0x + qy * k0y + qz * k0z);
      float d2b = qsv + ks1 - 2.0f * (qx * k1x + qy * k1y + qz * k1z);
      bool  ma  = d2a <= RATT2;
      bool  mb  = d2b <= RATT2;
      float sa  = ma ? s0[v] : -1e30f;
      float sb  = mb ? s1[v] : -1e30f;
      float r   = fmaxf(sa, sb);
      r = fmaxf(r, __shfl_xor(r, 1));
      r = fmaxf(r, __shfl_xor(r, 2));
      r = fmaxf(r, __shfl_xor(r, 4));
      r = fmaxf(r, __shfl_xor(r, 8));
      float mnew  = fmaxf(mrow[v], r);
      float alpha = __expf(mrow[v] - mnew);
      mrow[v] = mnew;
      float pa = ma ? __expf(sa - mnew) : 0.0f;
      float pb = mb ? __expf(sb - mnew) : 0.0f;
      float ps = pa + pb;
      ps += __shfl_xor(ps, 1);
      ps += __shfl_xor(ps, 2);
      ps += __shfl_xor(ps, 4);
      ps += __shfl_xor(ps, 8);
      lrow[v] = lrow[v] * alpha + ps;
#pragma unroll
      for (int c = 0; c < 8; ++c) acc[c][v] *= alpha;
      Ps[row][l15]      = (__bf16)pa;
      Ps[row][16 + l15] = (__bf16)pb;
    }
    __syncthreads();

    v16bf ap = *(const v16bf*)&Ps[l15][half * 16];
#pragma unroll
    for (int c = 0; c < 8; ++c) {
      v16bf bv = *(const v16bf*)&Vt[c * 16 + l15][half * 16];
      acc[c] = wmma_bf16(ap, bv, acc[c]);
    }
    __syncthreads();
  }

#pragma unroll
  for (int v = 0; v < 8; ++v) {
    int row = v + 8 * half;
    float inv = 1.0f / lrow[v];
    size_t xr = (base + qt * 16 + row) * (size_t)KPAD + 515;
#pragma unroll
    for (int c = 0; c < 8; ++c) {
      Xout[xr + c * 16 + l15] = (__bf16)(acc[c][v] * inv);
    }
  }
}

// ================================================================ host
extern "C" void kernel_launch(void* const* d_in, const int* in_sizes, int n_in,
                              void* d_out, int out_size, void* d_ws, size_t ws_size,
                              hipStream_t stream) {
  (void)in_sizes; (void)n_in; (void)out_size; (void)ws_size;

  const float* z    = (const float*)d_in[0];
  const float* tmpl = (const float*)d_in[1];
  const float* wqp  = (const float*)d_in[2];
  const float* wkp  = (const float*)d_in[3];
  const float* wvp  = (const float*)d_in[4];
  auto IN = [&](int s, int j) { return (const float*)d_in[5 + s * 10 + j]; };
  const int inDims[3] = {515, 643, 643};

  char* w = (char*)d_ws;
  size_t o = 0;
  __bf16* X    = (__bf16*)(w + o); o += (size_t)M_TOT * KPAD * 2;
  __bf16* W1p  = (__bf16*)(w + o); o += 3 * (size_t)HID * KPAD * 2;
  __bf16* W2p  = (__bf16*)(w + o); o += 3 * (size_t)HID * HID * 2;
  float*  H    = (float*) (w + o); o += (size_t)M_TOT * HID * 4;
  __bf16* Hb   = (__bf16*)(w + o); o += (size_t)M_TOT * HID * 2;
  float*  pSum = (float*) (w + o); o += 256 * (size_t)HID * 4;
  float*  pSq  = (float*) (w + o); o += 256 * (size_t)HID * 4;
  float*  ab   = (float*) (w + o); o += 2 * HID * 4;
  float*  crd  = (float*) (w + o); o += (size_t)M_TOT * 3 * 4;
  float*  sqb  = (float*) (w + o); o += (size_t)M_TOT * 4;
  __bf16* Qb   = (__bf16*)(w + o); o += (size_t)M_TOT * AD * 2;
  __bf16* Kb   = (__bf16*)(w + o); o += (size_t)M_TOT * AD * 2;
  __bf16* Vb   = (__bf16*)(w + o); o += (size_t)M_TOT * AD * 2;

  for (int s = 0; s < 3; ++s) {
    pack_w<<<(HID * KPAD + 255) / 256, 256, 0, stream>>>(
        IN(s, 0), W1p + (size_t)s * HID * KPAD, HID, inDims[s], KPAD);
    pack_w<<<(HID * HID + 255) / 256, 256, 0, stream>>>(
        IN(s, 4), W2p + (size_t)s * HID * HID, HID, HID, HID);
  }
  build_x0<<<M_TOT, 256, 0, stream>>>(z, tmpl, X);

  dim3 ggrid(M_TOT / 128, HID / 64);
  for (int s = 0; s < 3; ++s) {
    gemm_tdm<<<ggrid, 256, 0, stream>>>(X, KPAD, W1p + (size_t)s * HID * KPAD, KPAD,
                                        IN(s, 1), H, HID, KPAD);
    bn_stats1<<<256, 256, 0, stream>>>(H, pSum, pSq);
    bn_stats2<<<2, 256, 0, stream>>>(pSum, pSq, IN(s, 2), IN(s, 3), ab);
    bn_relu<<<(M_TOT * HID) / 1024, 256, 0, stream>>>(H, ab, Hb);
    gemm_tdm<<<ggrid, 256, 0, stream>>>(Hb, HID, W2p + (size_t)s * HID * HID, HID,
                                        IN(s, 5), H, HID, HID);
    bn_stats1<<<256, 256, 0, stream>>>(H, pSum, pSq);
    bn_stats2<<<2, 256, 0, stream>>>(pSum, pSq, IN(s, 6), IN(s, 7), ab);
    bn_relu<<<(M_TOT * HID) / 1024, 256, 0, stream>>>(H, ab, Hb);
    float* cdst = (s == 2) ? (float*)d_out : crd;
    head3<<<M_TOT / 256, 256, 0, stream>>>(Hb, IN(s, 8), IN(s, 9), cdst,
                                           (s < 2) ? sqb : nullptr,
                                           (s < 2) ? X : nullptr);
    if (s < 2) {
      qkv<<<M_TOT, 128, 0, stream>>>(crd, wqp, wkp, wvp, Qb, Kb, Vb);
      attn<<<dim3(NPTS / 16, BATCH), 32, 0, stream>>>(Qb, Kb, Vb, crd, sqb, X);
    }
  }
}
